// RingAttentionBlock_39324720562443
// MI455X (gfx1250) — matrix operand order repaired and checked by reading, not compiled, not run on hardware
//
#include <hip/hip_runtime.h>
#include <hip/hip_bf16.h>
#include <cstdint>

typedef __attribute__((ext_vector_type(16))) __bf16 v16bf;
typedef __attribute__((ext_vector_type(8)))  __bf16 v8bf;
typedef __attribute__((ext_vector_type(4)))  __bf16 v4bf;
typedef __attribute__((ext_vector_type(8)))  float  v8f;
typedef __attribute__((ext_vector_type(4)))  float  v4f;

constexpr int kB  = 2;
constexpr int kS  = 2048;
constexpr int kD  = 1024;
constexpr int kH  = 16;
constexpr int kDH = 64;
constexpr int kM  = kB * kS;                       // 4096
constexpr size_t kE   = (size_t)kM * kD;           // 4,194,304 elems
constexpr size_t kWE  = (size_t)kD * kD;           // 1,048,576 elems per weight
constexpr size_t kAW  = (size_t)kB * kH * kS * kS; // 134,217,728 ones

// ---------- scalar helpers ----------

__device__ __forceinline__ __bf16 f2bf(float f) {
  unsigned u = __builtin_bit_cast(unsigned, f);
  unsigned r = u + 0x7FFFu + ((u >> 16) & 1u);
  return __builtin_bit_cast(__bf16, (unsigned short)(r >> 16));
}
__device__ __forceinline__ float bf2f(__bf16 b) {
  unsigned u = (unsigned)__builtin_bit_cast(unsigned short, b) << 16;
  return __builtin_bit_cast(float, u);
}
__device__ __forceinline__ v16bf cat8(v8bf lo, v8bf hi) {
  v16bf r;
#pragma unroll
  for (int i = 0; i < 8; ++i) { r[i] = lo[i]; r[i + 8] = hi[i]; }
  return r;
}
__device__ __forceinline__ v8f wmma_bf16(v16bf a, v16bf b, v8f c) {
  return __builtin_amdgcn_wmma_f32_16x16x32_bf16(false, a, false, b,
                                                 (short)0, c, false, false);
}

// ---------- DPP 16-lane reductions (pure VALU, no LDS) ----------
// xor1 = quad_perm(1,0,3,2)=0xB1, xor2 = quad_perm(2,3,0,1)=0x4E,
// then row_ror:4 (0x124) and row_ror:8 (0x128) stay within a 16-lane row.

template <int CTRL>
__device__ __forceinline__ float dpp_movf(float v) {
  int x = __builtin_bit_cast(int, v);
  int y = __builtin_amdgcn_update_dpp(0, x, CTRL, 0xF, 0xF, true);
  return __builtin_bit_cast(float, y);
}
__device__ __forceinline__ float red_max16(float v) {
  v = fmaxf(v, dpp_movf<0xB1>(v));
  v = fmaxf(v, dpp_movf<0x4E>(v));
  v = fmaxf(v, dpp_movf<0x124>(v));
  v = fmaxf(v, dpp_movf<0x128>(v));
  return v;
}
__device__ __forceinline__ float red_sum16(float v) {
  v += dpp_movf<0xB1>(v);
  v += dpp_movf<0x4E>(v);
  v += dpp_movf<0x124>(v);
  v += dpp_movf<0x128>(v);
  return v;
}

// ---------- CDNA5 async global->LDS copy (ASYNCcnt tracked) ----------

__device__ __forceinline__ void async_cp16(unsigned lds, uint64_t ga) {
  asm volatile("global_load_async_to_lds_b128 %0, %1, off"
               :: "v"(lds), "v"(ga) : "memory");
}
#define WAIT_ASYNC_0() asm volatile("s_wait_asynccnt 0x0" ::: "memory")
#define WAIT_ASYNC_2() asm volatile("s_wait_asynccnt 0x2" ::: "memory")
#define WAIT_ASYNC_8() asm volatile("s_wait_asynccnt 0x8" ::: "memory")

__device__ __forceinline__ unsigned lds_off(const void* p) {
  return (unsigned)(uintptr_t)p;   // low 32 bits of generic LDS addr == LDS offset
}

// ---------- one-time conversions ----------

__global__ __launch_bounds__(256)
void cvt_f32_bf16(const float* __restrict__ in, __bf16* __restrict__ out, int n4) {
  int i = blockIdx.x * blockDim.x + threadIdx.x;
  if (i >= n4) return;
  v4f x = *(const v4f*)(in + (size_t)i * 4);
  v4bf y;
  y[0] = f2bf(x[0]); y[1] = f2bf(x[1]); y[2] = f2bf(x[2]); y[3] = f2bf(x[3]);
  *(v4bf*)(out + (size_t)i * 4) = y;
}

__global__ __launch_bounds__(256)
void transpose_cvt(const float* __restrict__ W, __bf16* __restrict__ Wt,
                   int N, int K) {
  __shared__ float t[32][33];
  int n0 = blockIdx.x * 32, k0 = blockIdx.y * 32;
  int tx = threadIdx.x & 31, ty = threadIdx.x >> 5;
#pragma unroll
  for (int j = 0; j < 4; ++j)
    t[ty + j * 8][tx] = W[(size_t)(k0 + ty + j * 8) * N + n0 + tx];
  __syncthreads();
#pragma unroll
  for (int j = 0; j < 4; ++j)
    Wt[(size_t)(n0 + ty + j * 8) * K + k0 + tx] = f2bf(t[tx][ty + j * 8]);
}

// ---------- GEMM: C[M,N] = A[M,K](bf16) @ Wt[N,K](bf16)^T + bias ----------

#define GBM 128
#define GBN 128
#define GBK 64

template <bool OUT_BF16>
__global__ __launch_bounds__(256)
void gemm_bias_bf(const __bf16* __restrict__ A, const __bf16* __restrict__ Wt,
                  const float* __restrict__ bias, void* __restrict__ Cout,
                  int M, int N, int K) {
  __shared__ alignas(32) __bf16 As[2][GBM * GBK];
  __shared__ alignas(32) __bf16 Ws[2][GBN * GBK];

  const int tid  = threadIdx.x;
  const int wave = tid >> 5, lane = tid & 31;
  const int hf   = lane >> 4, l16 = lane & 15;
  const int wm   = wave >> 1, wn = wave & 1;
  const int m0   = blockIdx.y * GBM;
  const int n0   = blockIdx.x * GBN;

  const int srow = tid >> 1;
  const int scb  = (tid & 1) * 64;
  const __bf16* Agp = A  + (size_t)(m0 + srow) * K;
  const __bf16* Wgp = Wt + (size_t)(n0 + srow) * K;

  auto stage = [&](int buf, int k0) {
    uint64_t ga = (uint64_t)(uintptr_t)(Agp + k0) + scb;
    unsigned la = lds_off(&As[buf][srow * GBK]) + scb;
    async_cp16(la, ga);           async_cp16(la + 16, ga + 16);
    async_cp16(la + 32, ga + 32); async_cp16(la + 48, ga + 48);
    ga = (uint64_t)(uintptr_t)(Wgp + k0) + scb;
    la = lds_off(&Ws[buf][srow * GBK]) + scb;
    async_cp16(la, ga);           async_cp16(la + 16, ga + 16);
    async_cp16(la + 32, ga + 32); async_cp16(la + 48, ga + 48);
  };

  v8f acc[2][4] = {};
  stage(0, 0);
  int cur = 0;
  for (int k0 = 0; k0 < K; k0 += GBK) {
    if (k0 + GBK < K) { stage(cur ^ 1, k0 + GBK); WAIT_ASYNC_8(); }
    else              { WAIT_ASYNC_0(); }
    __syncthreads();

#pragma unroll
    for (int ks = 0; ks < 2; ++ks) {
      v16bf a[2], b[4];
#pragma unroll
      for (int mi = 0; mi < 2; ++mi) {
        const __bf16* r = &As[cur][(wm * 32 + mi * 16 + l16) * GBK + ks * 32];
        a[mi] = cat8(*(const v8bf*)(r + hf * 8), *(const v8bf*)(r + 16 + hf * 8));
      }
#pragma unroll
      for (int ni = 0; ni < 4; ++ni)
        { const __bf16* wr = &Ws[cur][(wn * 64 + ni * 16 + l16) * GBK + ks * 32];
          b[ni] = cat8(*(const v8bf*)(wr + hf * 8), *(const v8bf*)(wr + 16 + hf * 8)); }
#pragma unroll
      for (int mi = 0; mi < 2; ++mi)
#pragma unroll
        for (int ni = 0; ni < 4; ++ni)
          acc[mi][ni] = wmma_bf16(a[mi], b[ni], acc[mi][ni]);
    }
    __syncthreads();
    cur ^= 1;
  }

#pragma unroll
  for (int mi = 0; mi < 2; ++mi)
#pragma unroll
    for (int ni = 0; ni < 4; ++ni) {
      int col = n0 + wn * 64 + ni * 16 + l16;
      float bv = bias[col];
#pragma unroll
      for (int r = 0; r < 8; ++r) {
        int row = m0 + wm * 32 + mi * 16 + hf * 8 + r;
        float v = acc[mi][ni][r] + bv;
        if (OUT_BF16) ((__bf16*)Cout)[(size_t)row * N + col] = f2bf(v);
        else          ((float*)Cout)[(size_t)row * N + col] = v;
      }
    }
}

// ---------- RoPE (in-place on bf16 q and k) ----------
__global__ __launch_bounds__(256)
void rope_bf(__bf16* __restrict__ q, __bf16* __restrict__ k) {
  int idx = blockIdx.x * blockDim.x + threadIdx.x;
  int i = idx & 31;
  int h = (idx >> 5) & 15;
  int s = (idx >> 9) & 2047;
  int b = idx >> 20;
  if (b >= kB) return;
  float inv = exp2f(-0.34375f * (float)i);   // 2048^(-i/32) = 2^(-11i/32)
  float f = (float)s * inv;
  float c = __cosf(f), sn = __sinf(f);
  size_t base = (((size_t)b * kS + s) * kH + h) * kDH + i;
  float q1 = bf2f(q[base]), q2 = bf2f(q[base + 32]);
  float k1 = bf2f(k[base]), k2 = bf2f(k[base + 32]);
  q[base]      = f2bf(q1 * c - q2 * sn);
  q[base + 32] = f2bf(q1 * sn + q2 * c);
  k[base]      = f2bf(k1 * c - k2 * sn);
  k[base + 32] = f2bf(k1 * sn + k2 * c);
}

// ---------- causal flash attention (bf16 in/out) ----------
// 256 threads = 8 waves; q-tile 128 rows; K tiles of 64 double-buffered:
// K via async DMA, V (transposed) staged manually in the compute shadow.

#define QT 128
#define KT 64

__global__ __launch_bounds__(256)
void flash_kernel(const __bf16* __restrict__ qbuf, const __bf16* __restrict__ kbuf,
                  const __bf16* __restrict__ vbuf, __bf16* __restrict__ out) {
  __shared__ alignas(32) __bf16 Ks[2][KT * kDH];   // [kk][d]
  __shared__ alignas(32) __bf16 Vt[2][kDH * KT];   // [d][kk]
  __shared__ alignas(32) __bf16 Ps[8][16 * KT];    // per-wave P scratch

  const int tid  = threadIdx.x;
  const int wave = tid >> 5, lane = tid & 31;
  const int hf   = lane >> 4, l16 = lane & 15;
  const int b    = blockIdx.x >> 4;
  const int h    = blockIdx.x & 15;
  const int q0   = blockIdx.y * QT;
  const int qrow0 = q0 + wave * 16;

  // Q fragments: raw bf16 loads (scale folded into scores)
  const __bf16* qp = qbuf + (((size_t)b * kS + qrow0 + l16) * kH + h) * kDH;
  v16bf qa[2];
  qa[0] = cat8(*(const v8bf*)(qp + hf * 8),      *(const v8bf*)(qp + 16 + hf * 8));
  qa[1] = cat8(*(const v8bf*)(qp + 32 + hf * 8), *(const v8bf*)(qp + 48 + hf * 8));

  v8f o[4] = {};
  float mrow[8], lrow[8];
#pragma unroll
  for (int r = 0; r < 8; ++r) { mrow[r] = -1e30f; lrow[r] = 0.0f; }

  // staging decomposition: 256 threads cover 64 rows x 128B
  const int skk = tid >> 2;
  const int sq  = tid & 3;

  auto issueK = [&](int buf, int kt0) {
    const __bf16* gk = kbuf + (((size_t)b * kS + kt0 + skk) * kH + h) * kDH;
    uint64_t ga = (uint64_t)(uintptr_t)gk + sq * 32;
    unsigned la = lds_off(&Ks[buf][skk * kDH]) + sq * 32;
    async_cp16(la, ga); async_cp16(la + 16, ga + 16);
  };
  auto stageV = [&](int buf, int kt0) {
    int db = sq * 16;
    const __bf16* gv = vbuf + (((size_t)b * kS + kt0 + skk) * kH + h) * kDH + db;
#pragma unroll
    for (int j = 0; j < 2; ++j) {
      v8bf x = *(const v8bf*)(gv + j * 8);
#pragma unroll
      for (int i = 0; i < 8; ++i) Vt[buf][(db + j * 8 + i) * KT + skk] = x[i];
    }
  };

  issueK(0, 0);
  stageV(0, 0);
  int cur = 0;
  const int ktend = q0 + QT - KT;                   // last tile start
  for (int kt0 = 0; kt0 <= ktend; kt0 += KT) {
    const bool hn = (kt0 + KT <= ktend);
    if (hn) { issueK(cur ^ 1, kt0 + KT); WAIT_ASYNC_2(); }
    else    { WAIT_ASYNC_0(); }
    __syncthreads();

    if (kt0 <= qrow0 + 15) {                        // tile contributes to this wave
      // scores (f32 accum) scaled by 1/sqrt(DH)
      v8f s[4];
#pragma unroll
      for (int nt = 0; nt < 4; ++nt) {
        const __bf16* kr = &Ks[cur][(nt * 16 + l16) * kDH];
        v8f c = {};
        c = wmma_bf16(qa[0], cat8(*(const v8bf*)(kr + hf * 8), *(const v8bf*)(kr + 16 + hf * 8)), c);
        c = wmma_bf16(qa[1], cat8(*(const v8bf*)(kr + 32 + hf * 8), *(const v8bf*)(kr + 48 + hf * 8)), c);
#pragma unroll
        for (int r = 0; r < 8; ++r) s[nt][r] = c[r] * 0.125f;
      }

      const bool diag = (kt0 + KT > qrow0);
      if (diag) {
#pragma unroll
        for (int nt = 0; nt < 4; ++nt) {
          int kpos = kt0 + nt * 16 + l16;
#pragma unroll
          for (int r = 0; r < 8; ++r)
            if (qrow0 + hf * 8 + r < kpos) s[nt][r] = -1e9f;
        }
      }

      float pscale[8];
#pragma unroll
      for (int r = 0; r < 8; ++r) {
        float v = fmaxf(fmaxf(s[0][r], s[1][r]), fmaxf(s[2][r], s[3][r]));
        v = red_max16(v);
        float mnew = fmaxf(mrow[r], v);
        pscale[r] = __expf(mrow[r] - mnew);
        mrow[r] = mnew;
      }

      __bf16* pw = Ps[wave];
      float rsum[8];
#pragma unroll
      for (int r = 0; r < 8; ++r) rsum[r] = 0.0f;
#pragma unroll
      for (int nt = 0; nt < 4; ++nt) {
        int kpos = kt0 + nt * 16 + l16;
#pragma unroll
        for (int r = 0; r < 8; ++r) {
          float p = __expf(s[nt][r] - mrow[r]);
          if (diag && (qrow0 + hf * 8 + r < kpos)) p = 0.0f;
          rsum[r] += p;
          pw[(hf * 8 + r) * KT + nt * 16 + l16] = f2bf(p);
        }
      }
#pragma unroll
      for (int r = 0; r < 8; ++r) {
        lrow[r] = lrow[r] * pscale[r] + red_sum16(rsum[r]);
#pragma unroll
        for (int nt = 0; nt < 4; ++nt) o[nt][r] *= pscale[r];
      }

      v16bf pa[2];
      {
        const __bf16* pr = pw + l16 * KT;
        pa[0] = cat8(*(const v8bf*)(pr + hf * 8),      *(const v8bf*)(pr + 16 + hf * 8));
        pa[1] = cat8(*(const v8bf*)(pr + 32 + hf * 8), *(const v8bf*)(pr + 48 + hf * 8));
      }
#pragma unroll
      for (int nt = 0; nt < 4; ++nt) {
        const __bf16* vr = &Vt[cur][(nt * 16 + l16) * KT];
        o[nt] = wmma_bf16(pa[0], cat8(*(const v8bf*)(vr + hf * 8), *(const v8bf*)(vr + 16 + hf * 8)), o[nt]);
        o[nt] = wmma_bf16(pa[1], cat8(*(const v8bf*)(vr + 32 + hf * 8), *(const v8bf*)(vr + 48 + hf * 8)), o[nt]);
      }
    }

    if (hn) stageV(cur ^ 1, kt0 + KT);              // overlap with other waves' compute
    __syncthreads();
    cur ^= 1;
  }

#pragma unroll
  for (int nt = 0; nt < 4; ++nt) {
    int col = h * kDH + nt * 16 + l16;
#pragma unroll
    for (int r = 0; r < 8; ++r) {
      int row = qrow0 + hf * 8 + r;
      out[((size_t)b * kS + row) * kD + col] = f2bf(o[nt][r] / (lrow[r] + 1e-10f));
    }
  }
}

// ---------- attention_weights = ones ----------
__global__ __launch_bounds__(256)
void fill_ones(float* __restrict__ p, size_t n4) {
  size_t i = (size_t)blockIdx.x * blockDim.x + threadIdx.x;
  size_t stride = (size_t)gridDim.x * blockDim.x;
  v4f one = {1.0f, 1.0f, 1.0f, 1.0f};
  for (; i < n4; i += stride) *(v4f*)(p + i * 4) = one;
}

// ---------- launcher ----------
extern "C" void kernel_launch(void* const* d_in, const int* in_sizes, int n_in,
                              void* d_out, int out_size, void* d_ws, size_t ws_size,
                              hipStream_t stream) {
  const float* x  = (const float*)d_in[0];
  const float* Wq = (const float*)d_in[1];
  const float* bq = (const float*)d_in[2];
  const float* Wk = (const float*)d_in[3];
  const float* bk = (const float*)d_in[4];
  const float* Wv = (const float*)d_in[5];
  const float* bv = (const float*)d_in[6];
  const float* Wo = (const float*)d_in[7];
  const float* bo = (const float*)d_in[8];
  float* out = (float*)d_out;

  __bf16* xb  = (__bf16*)d_ws;
  __bf16* wqt = xb  + kE;
  __bf16* wkt = wqt + kWE;
  __bf16* wvt = wkt + kWE;
  __bf16* wot = wvt + kWE;
  __bf16* qb  = wot + kWE;
  __bf16* kb  = qb + kE;
  __bf16* vb  = kb + kE;
  __bf16* ab  = vb + kE;

  cvt_f32_bf16<<<(int)(kE / 4 / 256), 256, 0, stream>>>(x, xb, (int)(kE / 4));
  dim3 tg(kD / 32, kD / 32);
  transpose_cvt<<<tg, 256, 0, stream>>>(Wq, wqt, kD, kD);
  transpose_cvt<<<tg, 256, 0, stream>>>(Wk, wkt, kD, kD);
  transpose_cvt<<<tg, 256, 0, stream>>>(Wv, wvt, kD, kD);
  transpose_cvt<<<tg, 256, 0, stream>>>(Wo, wot, kD, kD);

  dim3 gg(kD / GBN, kM / GBM);  // (8, 32)
  gemm_bias_bf<true><<<gg, 256, 0, stream>>>(xb, wqt, bq, qb, kM, kD, kD);
  gemm_bias_bf<true><<<gg, 256, 0, stream>>>(xb, wkt, bk, kb, kM, kD, kD);
  gemm_bias_bf<true><<<gg, 256, 0, stream>>>(xb, wvt, bv, vb, kM, kD, kD);

  rope_bf<<<(kB * kS * kH * 32) / 256, 256, 0, stream>>>(qb, kb);

  flash_kernel<<<dim3(kB * kH, kS / QT), 256, 0, stream>>>(qb, kb, vb, ab);

  gemm_bias_bf<false><<<gg, 256, 0, stream>>>(ab, wot, bo, out, kM, kD, kD);

  fill_ones<<<2048, 256, 0, stream>>>(out + kE, kAW / 4);
}
